// EGNN_1168231105097
// MI455X (gfx1250) — compile-verified, matrix operand order checked by reading, not run on hardware
//
#include <hip/hip_runtime.h>
#include <hip/hip_bf16.h>

#define N_NODES   20000
#define N_EDGES   320000
#define HID       128
#define NUM_IN    11
#define PE_IN     24
#define NUM_LAYERS 3
#define NUM_GRAPHS 64

typedef __attribute__((ext_vector_type(16))) __bf16 bf16x16;
typedef __attribute__((ext_vector_type(8)))  float  floatx8;

struct U32x8 { uint4 a, b; };

static __device__ __forceinline__ bf16x16 make_frag(uint4 a, uint4 b) {
    U32x8 t{a, b};
    return __builtin_bit_cast(bf16x16, t);
}

// round-to-nearest-even f32 -> bf16 (as raw u16)
static __device__ __forceinline__ unsigned short f2bf(float f) {
    unsigned int u = __float_as_uint(f);
    u += 0x7FFFu + ((u >> 16) & 1u);
    return (unsigned short)(u >> 16);
}

static __device__ __forceinline__ floatx8 zero8() {
    floatx8 z = {0.f, 0.f, 0.f, 0.f, 0.f, 0.f, 0.f, 0.f};
    return z;
}

// A fragment: lane covers row m, K = {kBase..kBase+7} ++ {kBase+16..kBase+23}
static __device__ __forceinline__ bf16x16 load_frag_a(const unsigned short* rowp, int col) {
    uint4 a = *reinterpret_cast<const uint4*>(rowp + col);
    uint4 b = *reinterpret_cast<const uint4*>(rowp + col + 16);
    return make_frag(a, b);
}

// B fragment: prepacked, 16 contiguous bf16 per lane
static __device__ __forceinline__ bf16x16 load_frag_b(const unsigned short* p) {
    const uint4* q = reinterpret_cast<const uint4*>(p);
    return make_frag(q[0], q[1]);
}

static __device__ __forceinline__ floatx8 wmma_bf16(bf16x16 a, bf16x16 b, floatx8 c) {
    return __builtin_amdgcn_wmma_f32_16x16x32_bf16(false, a, false, b, (short)0, c, false, false);
}

// Native single-op L2 f32 atomic add (global_atomic_add_f32, no return):
// relaxed + agent scope avoids any CAS loop fallback.
static __device__ __forceinline__ void atomic_add_f32(float* p, float v) {
#if defined(__has_builtin) && __has_builtin(__hip_atomic_fetch_add)
    __hip_atomic_fetch_add(p, v, __ATOMIC_RELAXED, __HIP_MEMORY_SCOPE_AGENT);
#else
    atomicAdd(p, v);
#endif
}

template <int ACT>
static __device__ __forceinline__ float act(float v) {
    if constexpr (ACT == 0) return v / (1.f + __expf(-v));   // silu
    else                    return tanhf(v);                  // tanh
}

// ---------------------------------------------------------------------------
// Weight prepack: [K,128] f32 row-major -> WMMA B-operand bf16 tile layout.
// Tile (kt,nt): lane l holds n = nt*16 + (l&15); kBase = (l>>4)*8;
// elem j: k = kt*32 + kBase + (j&7) + ((j>>3)<<4)
// ---------------------------------------------------------------------------
__global__ void k_prepack(const float* __restrict__ w, unsigned short* __restrict__ dst, int K) {
    int o = blockIdx.x * blockDim.x + threadIdx.x;
    int total = K * HID;
    if (o >= total) return;
    int tile   = o >> 9;
    int within = o & 511;
    int lane = within >> 4;
    int j    = within & 15;
    int kt = tile >> 3;     // NT = 128/16 = 8
    int nt = tile & 7;
    int n     = nt * 16 + (lane & 15);
    int kBase = (lane >> 4) * 8;
    int k = kt * 32 + kBase + (j & 7) + ((j >> 3) << 4);
    dst[o] = f2bf(w[(size_t)k * HID + n]);
}

// ---------------------------------------------------------------------------
// Node embeddings (K=11 / K=24: too shallow for WMMA, pure VALU)
// ---------------------------------------------------------------------------
__global__ void k_embed(const float* __restrict__ x_in, const float* __restrict__ rw,
                        const float* __restrict__ ew, const float* __restrict__ eb,
                        const float* __restrict__ pw, const float* __restrict__ pb,
                        float* __restrict__ xh, unsigned short* __restrict__ x_bf,
                        float* __restrict__ pe, unsigned short* __restrict__ pe_bf) {
    int node = blockIdx.x;
    int c    = threadIdx.x;
    float vx = eb[c];
#pragma unroll
    for (int k = 0; k < NUM_IN; ++k) vx += x_in[node * NUM_IN + k] * ew[k * HID + c];
    float vp = pb[c];
#pragma unroll
    for (int k = 0; k < PE_IN; ++k) vp += rw[node * PE_IN + k] * pw[k * HID + c];
    size_t idx = (size_t)node * HID + c;
    xh[idx] = vx; x_bf[idx] = f2bf(vx);
    pe[idx] = vp; pe_bf[idx] = f2bf(vp);
}

__global__ void k_dist(const int* __restrict__ send, const int* __restrict__ rec,
                       const float* __restrict__ pos, float* __restrict__ dist) {
    int e = blockIdx.x * blockDim.x + threadIdx.x;
    if (e >= N_EDGES) return;
    int s = send[e], r = rec[e];
    float dx = pos[s * 3 + 0] - pos[r * 3 + 0];
    float dy = pos[s * 3 + 1] - pos[r * 3 + 1];
    float dz = pos[s * 3 + 2] - pos[r * 3 + 2];
    dist[e] = sqrtf(dx * dx + dy * dy + dz * dz);
}

__global__ void k_zero(float* __restrict__ p, int n) {
    int i = blockIdx.x * blockDim.x + threadIdx.x;
    if (i < n) p[i] = 0.f;
}

__global__ void k_cvt(const float* __restrict__ src, unsigned short* __restrict__ dst, int n) {
    int i = blockIdx.x * blockDim.x + threadIdx.x;
    if (i < n) dst[i] = f2bf(src[i]);
}

// ---------------------------------------------------------------------------
// Edge MLP, dual-tile: one wave32 handles 32 edges (two 16-row tiles) so every
// B fragment load feeds two independent back-to-back WMMAs (co-executable,
// no D->A/B RAW hazard), halving weight-stream traffic.
// out = act(W2 @ act(W1 @ [srcA[idxA], srcB[idxB], dist] + b1) + b2)
// scatter-added into aggr[idxOut] with native f32 atomics.
// ---------------------------------------------------------------------------
template <int ACT>
__global__ void __launch_bounds__(32)
k_edge_mlp(const unsigned short* __restrict__ srcA, const int* __restrict__ idxA,
           const unsigned short* __restrict__ srcB, const int* __restrict__ idxB,
           const int* __restrict__ idxOut, const float* __restrict__ dist,
           const unsigned short* __restrict__ ppW1, const float* __restrict__ wdist,
           const float* __restrict__ b1,
           const unsigned short* __restrict__ ppW2, const float* __restrict__ b2,
           float* __restrict__ aggr) {
    const int lane  = threadIdx.x;
    const int m     = lane & 15;
    const int hi    = lane >> 4;
    const int kBase = hi * 8;
    const int base0 = blockIdx.x * 32;
    const int base1 = base0 + 16;

    // Per-lane gathers of A fragments (K = 256 over two 128-col sources)
    const unsigned short* pA0 = srcA + (size_t)idxA[base0 + m] * HID;
    const unsigned short* pB0 = srcB + (size_t)idxB[base0 + m] * HID;
    const unsigned short* pA1 = srcA + (size_t)idxA[base1 + m] * HID;
    const unsigned short* pB1 = srcB + (size_t)idxB[base1 + m] * HID;
    bf16x16 a0[8], a1[8];
#pragma unroll
    for (int kt = 0; kt < 4; ++kt) {
        a0[kt]     = load_frag_a(pA0, kt * 32 + kBase);
        a0[4 + kt] = load_frag_a(pB0, kt * 32 + kBase);
        a1[kt]     = load_frag_a(pA1, kt * 32 + kBase);
        a1[4 + kt] = load_frag_a(pB1, kt * 32 + kBase);
    }

    // D-layout epilogue row data: rows r + 8*hi within each tile
    float dd0[8], dd1[8];
    int   or0[8], or1[8];
#pragma unroll
    for (int r = 0; r < 8; ++r) {
        int e0 = base0 + hi * 8 + r;
        int e1 = base1 + hi * 8 + r;
        dd0[r] = dist[e0];  or0[r] = idxOut[e0];
        dd1[r] = dist[e1];  or1[r] = idxOut[e1];
    }

    __shared__ unsigned short hbuf[32 * HID];   // 8 KB intermediate

    // GEMM1: [32,256] x [256,128]  + rank-1 dist column + bias + activation
#pragma unroll
    for (int nt = 0; nt < 8; ++nt) {
        floatx8 acc0 = zero8(), acc1 = zero8();
#pragma unroll
        for (int kt = 0; kt < 8; ++kt) {
            bf16x16 bf = load_frag_b(ppW1 + ((size_t)(kt * 8 + nt) * 512 + lane * 16));
            acc0 = wmma_bf16(a0[kt], bf, acc0);
            acc1 = wmma_bf16(a1[kt], bf, acc1);
        }
        int n = nt * 16 + m;
        float bias = b1[n], wd = wdist[n];
#pragma unroll
        for (int r = 0; r < 8; ++r) {
            hbuf[(hi * 8 + r) * HID + n]        = f2bf(act<ACT>(acc0[r] + bias + dd0[r] * wd));
            hbuf[(16 + hi * 8 + r) * HID + n]   = f2bf(act<ACT>(acc1[r] + bias + dd1[r] * wd));
        }
    }
    __syncthreads();

    // GEMM2: [32,128] x [128,128]
    bf16x16 h0[4], h1f[4];
#pragma unroll
    for (int kt = 0; kt < 4; ++kt) {
        uint4 x0a = *reinterpret_cast<const uint4*>(&hbuf[m * HID + kt * 32 + kBase]);
        uint4 x0b = *reinterpret_cast<const uint4*>(&hbuf[m * HID + kt * 32 + kBase + 16]);
        h0[kt] = make_frag(x0a, x0b);
        uint4 x1a = *reinterpret_cast<const uint4*>(&hbuf[(16 + m) * HID + kt * 32 + kBase]);
        uint4 x1b = *reinterpret_cast<const uint4*>(&hbuf[(16 + m) * HID + kt * 32 + kBase + 16]);
        h1f[kt] = make_frag(x1a, x1b);
    }
#pragma unroll
    for (int nt = 0; nt < 8; ++nt) {
        floatx8 acc0 = zero8(), acc1 = zero8();
#pragma unroll
        for (int kt = 0; kt < 4; ++kt) {
            bf16x16 bf = load_frag_b(ppW2 + ((size_t)(kt * 8 + nt) * 512 + lane * 16));
            acc0 = wmma_bf16(h0[kt], bf, acc0);
            acc1 = wmma_bf16(h1f[kt], bf, acc1);
        }
        int n = nt * 16 + m;
        float bias = b2[n];
#pragma unroll
        for (int r = 0; r < 8; ++r) {
            atomic_add_f32(&aggr[(size_t)or0[r] * HID + n], act<ACT>(acc0[r] + bias));
            atomic_add_f32(&aggr[(size_t)or1[r] * HID + n], act<ACT>(acc1[r] + bias));
        }
    }
}

// ---------------------------------------------------------------------------
// Node MLP: layer2(ACT1(layer1([s0|s1|s2]) )) with selectable output handling.
// OUTMODE 0: io += v,  write io & bf16(io)          (x residual update)
// OUTMODE 1: io += tanh(v), write io & bf16(io)     (pe residual update)
// OUTMODE 2: atomicAdd(pooled[batch[row]], v)       (pre + segment-sum pool)
// ---------------------------------------------------------------------------
template <int K1T, int ACT1, int OUTMODE>
__global__ void __launch_bounds__(32)
k_node_mlp(const unsigned short* __restrict__ s0, const unsigned short* __restrict__ s1,
           const unsigned short* __restrict__ s2,
           const unsigned short* __restrict__ ppW1, const float* __restrict__ b1,
           const unsigned short* __restrict__ ppW2, const float* __restrict__ b2,
           float* __restrict__ io, unsigned short* __restrict__ out_bf,
           const int* __restrict__ batch, float* __restrict__ pooled) {
    const int lane  = threadIdx.x;
    const int m     = lane & 15;
    const int hi    = lane >> 4;
    const int kBase = hi * 8;
    const int base  = blockIdx.x * 16;
    const int row   = base + m;

    bf16x16 afrag[K1T];
#pragma unroll
    for (int kt = 0; kt < K1T; ++kt) {
        const unsigned short* sp = (kt < 4) ? s0 : ((kt < 8) ? s1 : s2);
        afrag[kt] = load_frag_a(sp + (size_t)row * HID, (kt & 3) * 32 + kBase);
    }

    __shared__ unsigned short hbuf[16 * HID];

#pragma unroll
    for (int nt = 0; nt < 8; ++nt) {
        floatx8 acc = zero8();
#pragma unroll
        for (int kt = 0; kt < K1T; ++kt)
            acc = wmma_bf16(afrag[kt],
                            load_frag_b(ppW1 + ((size_t)(kt * 8 + nt) * 512 + lane * 16)), acc);
        int n = nt * 16 + m;
        float bias = b1[n];
#pragma unroll
        for (int r = 0; r < 8; ++r)
            hbuf[(hi * 8 + r) * HID + n] = f2bf(act<ACT1>(acc[r] + bias));
    }
    __syncthreads();

    bf16x16 hfrag[4];
#pragma unroll
    for (int kt = 0; kt < 4; ++kt) {
        uint4 a = *reinterpret_cast<const uint4*>(&hbuf[m * HID + kt * 32 + kBase]);
        uint4 b = *reinterpret_cast<const uint4*>(&hbuf[m * HID + kt * 32 + kBase + 16]);
        hfrag[kt] = make_frag(a, b);
    }
#pragma unroll
    for (int nt = 0; nt < 8; ++nt) {
        floatx8 acc = zero8();
#pragma unroll
        for (int kt = 0; kt < 4; ++kt)
            acc = wmma_bf16(hfrag[kt],
                            load_frag_b(ppW2 + ((size_t)(kt * 8 + nt) * 512 + lane * 16)), acc);
        int n = nt * 16 + m;
        float bias = b2[n];
#pragma unroll
        for (int r = 0; r < 8; ++r) {
            int   grow = base + hi * 8 + r;
            float v    = acc[r] + bias;
            if constexpr (OUTMODE == 0) {
                size_t o  = (size_t)grow * HID + n;
                float  nv = io[o] + v;
                io[o] = nv;
                out_bf[o] = f2bf(nv);
            } else if constexpr (OUTMODE == 1) {
                size_t o  = (size_t)grow * HID + n;
                float  nv = io[o] + tanhf(v);
                io[o] = nv;
                out_bf[o] = f2bf(nv);
            } else {
                int g = batch[grow];
                atomic_add_f32(&pooled[(size_t)g * HID + n], v);
            }
        }
    }
}

// ---------------------------------------------------------------------------
// Readout: out[g] = ro2( silu( ro1(pooled[g]) ) )  — 1M MACs, scalar is fine
// ---------------------------------------------------------------------------
__global__ void k_readout(const float* __restrict__ pooled,
                          const float* __restrict__ w1, const float* __restrict__ b1,
                          const float* __restrict__ w2, const float* __restrict__ b2,
                          float* __restrict__ out) {
    int g = threadIdx.x;
    if (g >= NUM_GRAPHS) return;
    float s = 0.f;
    for (int j = 0; j < HID; ++j) {
        float h = b1[j];
        for (int k = 0; k < HID; ++k) h += pooled[g * HID + k] * w1[k * HID + j];
        h = h / (1.f + __expf(-h));
        s += h * w2[j];
    }
    out[g] = s + b2[0];
}

// ---------------------------------------------------------------------------
extern "C" void kernel_launch(void* const* d_in, const int* in_sizes, int n_in,
                              void* d_out, int out_size, void* d_ws, size_t ws_size,
                              hipStream_t stream) {
    (void)in_sizes; (void)n_in; (void)out_size; (void)ws_size;

    // ---- inputs (setup_inputs dict order, params flattened w,b recursively)
    const float* x_in = (const float*)d_in[0];
    const float* pos  = (const float*)d_in[1];
    const int*   eidx = (const int*)d_in[2];
    const int*   send = eidx;
    const int*   rec  = eidx + N_EDGES;
    const int*   batch = (const int*)d_in[3];
    const float* rw   = (const float*)d_in[4];

    int pi = 5;
    auto nf = [&]() { return (const float*)d_in[pi++]; };
    const float *embW = nf(), *embB = nf();
    const float *peW = nf(),  *peB  = nf();
    const float *pre1w = nf(), *pre1b = nf();
    const float *pre2w = nf(), *pre2b = nf();
    const float *ro1w = nf(),  *ro1b  = nf();
    const float *ro2w = nf(),  *ro2b  = nf();
    struct LayerP {
        const float *m1w, *m1b, *m2w, *m2b, *mp1w, *mp1b, *mp2w, *mp2b;
        const float *u1w, *u1b, *u2w, *u2b, *p1w, *p1b, *p2w, *p2b;
    } L[NUM_LAYERS];
    for (int l = 0; l < NUM_LAYERS; ++l) {
        L[l].m1w = nf(); L[l].m1b = nf(); L[l].m2w = nf(); L[l].m2b = nf();
        L[l].mp1w = nf(); L[l].mp1b = nf(); L[l].mp2w = nf(); L[l].mp2b = nf();
        L[l].u1w = nf(); L[l].u1b = nf(); L[l].u2w = nf(); L[l].u2b = nf();
        L[l].p1w = nf(); L[l].p1b = nf(); L[l].p2w = nf(); L[l].p2b = nf();
    }

    // ---- workspace carve (~65 MB)
    size_t off = 0;
    auto carve = [&](size_t bytes) -> void* {
        off = (off + 255) & ~(size_t)255;
        void* p = (char*)d_ws + off;
        off += bytes;
        return p;
    };
    float* xh   = (float*)carve((size_t)N_NODES * HID * 4);
    float* peh  = (float*)carve((size_t)N_NODES * HID * 4);
    unsigned short* x_bf  = (unsigned short*)carve((size_t)N_NODES * HID * 2);
    unsigned short* pe_bf = (unsigned short*)carve((size_t)N_NODES * HID * 2);
    float* aggr  = (float*)carve((size_t)N_NODES * HID * 4);
    float* aggrp = (float*)carve((size_t)N_NODES * HID * 4);
    unsigned short* aggr_bf  = (unsigned short*)carve((size_t)N_NODES * HID * 2);
    unsigned short* aggrp_bf = (unsigned short*)carve((size_t)N_NODES * HID * 2);
    float* dist   = (float*)carve((size_t)N_EDGES * 4);
    float* pooled = (float*)carve((size_t)NUM_GRAPHS * HID * 4);

    auto carve_pp = [&](int K) { return (unsigned short*)carve((size_t)K * HID * 2); };
    unsigned short *pp_m1[NUM_LAYERS], *pp_m2[NUM_LAYERS], *pp_mp1[NUM_LAYERS],
                   *pp_mp2[NUM_LAYERS], *pp_u1[NUM_LAYERS], *pp_u2[NUM_LAYERS],
                   *pp_p1[NUM_LAYERS], *pp_p2[NUM_LAYERS];
    for (int l = 0; l < NUM_LAYERS; ++l) {
        pp_m1[l] = carve_pp(256); pp_m2[l] = carve_pp(128);
        pp_mp1[l] = carve_pp(256); pp_mp2[l] = carve_pp(128);
        pp_u1[l] = carve_pp(384); pp_u2[l] = carve_pp(128);
        pp_p1[l] = carve_pp(256); pp_p2[l] = carve_pp(128);
    }
    unsigned short* pp_pre1 = carve_pp(128);
    unsigned short* pp_pre2 = carve_pp(128);

    auto prepack = [&](const float* w, unsigned short* dst, int K) {
        int total = K * HID;
        k_prepack<<<(total + 255) / 256, 256, 0, stream>>>(w, dst, K);
    };

    // ---- weight prepack (bf16 B-operand layout)
    for (int l = 0; l < NUM_LAYERS; ++l) {
        prepack(L[l].m1w, pp_m1[l], 256);   prepack(L[l].m2w, pp_m2[l], 128);
        prepack(L[l].mp1w, pp_mp1[l], 256); prepack(L[l].mp2w, pp_mp2[l], 128);
        prepack(L[l].u1w, pp_u1[l], 384);   prepack(L[l].u2w, pp_u2[l], 128);
        prepack(L[l].p1w, pp_p1[l], 256);   prepack(L[l].p2w, pp_p2[l], 128);
    }
    prepack(pre1w, pp_pre1, 128);
    prepack(pre2w, pp_pre2, 128);

    // ---- embeddings + edge distances
    k_embed<<<N_NODES, HID, 0, stream>>>(x_in, rw, embW, embB, peW, peB, xh, x_bf, peh, pe_bf);
    k_dist<<<(N_EDGES + 255) / 256, 256, 0, stream>>>(send, rec, pos, dist);

    const int nNH = N_NODES * HID;
    const int edgeGrid = N_EDGES / 32;   // 10000 (dual-tile waves)
    const int nodeGrid = N_NODES / 16;   // 1250

    // ---- message-passing layers
    for (int l = 0; l < NUM_LAYERS; ++l) {
        k_zero<<<(nNH + 255) / 256, 256, 0, stream>>>(aggr, nNH);
        k_zero<<<(nNH + 255) / 256, 256, 0, stream>>>(aggrp, nNH);

        // message = silu(msg2(silu(msg1([x[rec], pe[rec], dist]))))
        k_edge_mlp<0><<<edgeGrid, 32, 0, stream>>>(
            x_bf, rec, pe_bf, rec, rec, dist,
            pp_m1[l], L[l].m1w + (size_t)256 * HID, L[l].m1b,
            pp_m2[l], L[l].m2b, aggr);

        // message_pos = tanh(msgp2(tanh(msgp1([pe[send], pe[rec], dist]))))
        k_edge_mlp<1><<<edgeGrid, 32, 0, stream>>>(
            pe_bf, send, pe_bf, rec, rec, dist,
            pp_mp1[l], L[l].mp1w + (size_t)256 * HID, L[l].mp1b,
            pp_mp2[l], L[l].mp2b, aggrp);

        k_cvt<<<(nNH + 255) / 256, 256, 0, stream>>>(aggr, aggr_bf, nNH);
        k_cvt<<<(nNH + 255) / 256, 256, 0, stream>>>(aggrp, aggrp_bf, nNH);

        // x += upd2(silu(upd1([x, pe, aggr])))
        k_node_mlp<12, 0, 0><<<nodeGrid, 32, 0, stream>>>(
            x_bf, pe_bf, aggr_bf, pp_u1[l], L[l].u1b, pp_u2[l], L[l].u2b,
            xh, x_bf, nullptr, nullptr);

        // pe += tanh(pos2(tanh(pos1([pe, aggr_pos]))))
        k_node_mlp<8, 1, 1><<<nodeGrid, 32, 0, stream>>>(
            pe_bf, aggrp_bf, nullptr, pp_p1[l], L[l].p1b, pp_p2[l], L[l].p2b,
            peh, pe_bf, nullptr, nullptr);
    }

    // ---- pre-MLP + segment-sum pooling + readout
    k_zero<<<(NUM_GRAPHS * HID + 255) / 256, 256, 0, stream>>>(pooled, NUM_GRAPHS * HID);
    k_node_mlp<4, 0, 2><<<nodeGrid, 32, 0, stream>>>(
        x_bf, nullptr, nullptr, pp_pre1, pre1b, pp_pre2, pre2b,
        nullptr, nullptr, batch, pooled);

    k_readout<<<1, 64, 0, stream>>>(pooled, ro1w, ro1b, ro2w, ro2b, (float*)d_out);
}
